// SDTHead_73151882985662
// MI455X (gfx1250) — matrix operand order repaired and checked by reading, not compiled, not run on hardware
//
#include <hip/hip_runtime.h>
#include <math.h>

// ---------------------------------------------------------------------------
// MI455X (gfx1250) implementation. Compute-bound stages (fusion projection,
// two 3x3 convs) run as bf16 WMMA GEMMs (v_wmma_f32_16x16x32_bf16, wave32,
// 128x128 block tiles). Tile staging uses CDNA5 async global->LDS copies
// (global_load_async_to_lds_b128, ASYNCcnt) with a double-buffered LDS
// pipeline so loads for step kb+1 overlap WMMA on step kb.
// ---------------------------------------------------------------------------

typedef __bf16 bfx8  __attribute__((ext_vector_type(8)));
typedef __bf16 bfx16 __attribute__((ext_vector_type(16)));
typedef float  fx8   __attribute__((ext_vector_type(8)));

#define LDA 40   // 32 K-elems + 8 pad (bf16) per LDS row
#define LDB 40

__device__ __forceinline__ fx8 zero_fx8() {
    fx8 z;
#pragma unroll
    for (int i = 0; i < 8; ++i) z[i] = 0.0f;
    return z;
}

__device__ __forceinline__ bfx8 zero_bfx8() {
    bfx8 z;
#pragma unroll
    for (int i = 0; i < 8; ++i) z[i] = (__bf16)0.0f;
    return z;
}

// CDNA5 async copy: 16B global -> LDS, tracked by ASYNCcnt (per-lane).
// LDS aperture flat addresses carry the LDS byte offset in addr[31:0].
__device__ __forceinline__ void async_ld_b128(const void* gsrc, void* ldsdst) {
    unsigned lds = (unsigned)(size_t)ldsdst;
    unsigned long long g = (unsigned long long)(size_t)gsrc;
    asm volatile("global_load_async_to_lds_b128 %0, %1, off"
                 :: "v"(lds), "v"(g) : "memory");
}
__device__ __forceinline__ void wait_async0() {
    asm volatile("s_wait_asynccnt 0" ::: "memory");
}

// Build a v16bf A/B fragment from two 16B LDS chunks.
__device__ __forceinline__ bfx16 make_frag(const __bf16* p0, const __bf16* p1) {
    bfx8 lo = *(const bfx8*)p0;
    bfx8 hi = *(const bfx8*)p1;
    bfx16 r;
#pragma unroll
    for (int i = 0; i < 8; ++i) { r[i] = lo[i]; r[i + 8] = hi[i]; }
    return r;
}

// ---------------------------------------------------------------------------
// [C][N] f32 -> [N][C] bf16 transpose (per blockIdx.z batch slice).
// ---------------------------------------------------------------------------
__global__ __launch_bounds__(256)
void k_transpose_bf16(const float* __restrict__ in, __bf16* __restrict__ out,
                      int C, int N) {
    __shared__ float tile[32][33];
    const int b = blockIdx.z;
    in  += (size_t)b * C * N;
    out += (size_t)b * N * C;
    const int n0 = blockIdx.x * 32;
    const int c0 = blockIdx.y * 32;
    const int tx = threadIdx.x & 31;
    const int ty = threadIdx.x >> 5;
#pragma unroll
    for (int i = 0; i < 4; ++i) {
        int c = c0 + ty + i * 8;
        tile[ty + i * 8][tx] = in[(size_t)c * N + n0 + tx];
    }
    __syncthreads();
#pragma unroll
    for (int i = 0; i < 4; ++i) {
        int n = n0 + ty + i * 8;
        out[(size_t)n * C + c0 + tx] = (__bf16)tile[tx][ty + i * 8];
    }
}

// conv weight (O=256, I=256, 3,3) f32 -> [o][(ky*3+kx)*256 + i] bf16
__global__ __launch_bounds__(256)
void k_reshape_convw(const float* __restrict__ w, __bf16* __restrict__ wt, int total) {
    int idx = blockIdx.x * 256 + threadIdx.x;
    if (idx >= total) return;
    int o   = idx / 2304;
    int k   = idx - o * 2304;
    int tap = k >> 8;          // k / 256
    int ci  = k & 255;
    wt[idx] = (__bf16)w[((size_t)(o * 256 + ci)) * 9 + tap];
}

// ---------------------------------------------------------------------------
// Fusion: for each layer l, GEMM tokens[l] (4096x768) x projW[l]^T (256x768),
// exact GELU, softmax(layer_weights)-weighted sum.  Output x0 f32 NHWC.
// Grid: (4096/128, 256/128) = (32, 2); 256 threads = 8 waves (4 M x 2 N).
// ---------------------------------------------------------------------------
__global__ __launch_bounds__(256)
void k_fusion_wmma(const __bf16* __restrict__ featsT,  // [4][4096][768]
                   const __bf16* __restrict__ pwT,     // [4][256][768]
                   const float* __restrict__ lwRaw,    // [4]
                   float* __restrict__ out)            // [4096][256]
{
    __shared__ __bf16 Asl[2][128 * LDA];
    __shared__ __bf16 Bsl[2][128 * LDB];

    // softmax over 4 layer weights (cheap, every thread)
    float lw[4];
    {
        float mx = fmaxf(fmaxf(lwRaw[0], lwRaw[1]), fmaxf(lwRaw[2], lwRaw[3]));
        float s = 0.0f;
#pragma unroll
        for (int i = 0; i < 4; ++i) { lw[i] = __expf(lwRaw[i] - mx); s += lw[i]; }
        float inv = 1.0f / s;
#pragma unroll
        for (int i = 0; i < 4; ++i) lw[i] *= inv;
    }

    const int m0 = blockIdx.x * 128;
    const int n0 = blockIdx.y * 128;
    const int t  = threadIdx.x;
    const int lane = t & 31;
    const int wv = t >> 5;
    const int wm = wv >> 1;      // 0..3
    const int wn = wv & 1;       // 0..1
    const int mm = t >> 1;       // loader row 0..127
    const int kh = (t & 1) * 16; // loader K half

    fx8 facc[2][4];
#pragma unroll
    for (int i = 0; i < 2; ++i)
#pragma unroll
        for (int j = 0; j < 4; ++j) facc[i][j] = zero_fx8();

    auto load_tile = [&](int l, int kb, int buf) {
        const __bf16* sa = featsT + ((size_t)l * 4096 + m0 + mm) * 768 + (kb << 5) + kh;
        __bf16* da = &Asl[buf][mm * LDA + kh];
        async_ld_b128(sa,     da);
        async_ld_b128(sa + 8, da + 8);
        const __bf16* sb = pwT + ((size_t)l * 256 + n0 + mm) * 768 + (kb << 5) + kh;
        __bf16* db = &Bsl[buf][mm * LDB + kh];
        async_ld_b128(sb,     db);
        async_ld_b128(sb + 8, db + 8);
    };

    for (int l = 0; l < 4; ++l) {
        fx8 acc[2][4];
#pragma unroll
        for (int i = 0; i < 2; ++i)
#pragma unroll
            for (int j = 0; j < 4; ++j) acc[i][j] = zero_fx8();

        load_tile(l, 0, 0);
        wait_async0();
        __syncthreads();

        for (int kb = 0; kb < 24; ++kb) {
            const int cur = kb & 1;
            if (kb + 1 < 24) load_tile(l, kb + 1, 1 - cur);

            const int l16   = lane & 15;
            const int khalf = lane >> 4;
            bfx16 af[2], bfr[4];
#pragma unroll
            for (int mt = 0; mt < 2; ++mt) {
                const __bf16* p = &Asl[cur][(wm * 32 + mt * 16 + l16) * LDA];
                af[mt] = make_frag(p + khalf * 8, p + khalf * 8 + 16);
            }
#pragma unroll
            for (int nt = 0; nt < 4; ++nt) {
                const __bf16* p = &Bsl[cur][(wn * 64 + nt * 16 + l16) * LDB + khalf * 8];
                bfr[nt] = make_frag(p, p + 16);
            }
#pragma unroll
            for (int mt = 0; mt < 2; ++mt)
#pragma unroll
                for (int nt = 0; nt < 4; ++nt)
                    acc[mt][nt] = __builtin_amdgcn_wmma_f32_16x16x32_bf16(
                        false, af[mt], false, bfr[nt], (short)0, acc[mt][nt],
                        false, false);
            wait_async0();
            __syncthreads();
        }

        // exact GELU, weighted accumulate
        const float wl = lw[l];
#pragma unroll
        for (int mt = 0; mt < 2; ++mt)
#pragma unroll
            for (int nt = 0; nt < 4; ++nt)
#pragma unroll
                for (int v = 0; v < 8; ++v) {
                    float xv = acc[mt][nt][v];
                    float ge = 0.5f * xv * (1.0f + erff(xv * 0.70710678118654752f));
                    facc[mt][nt][v] += wl * ge;
                }
        __syncthreads();
    }

    // store f32 NHWC
#pragma unroll
    for (int mt = 0; mt < 2; ++mt)
#pragma unroll
        for (int nt = 0; nt < 4; ++nt) {
            const int col = n0 + wn * 64 + nt * 16 + (lane & 15);
#pragma unroll
            for (int v = 0; v < 8; ++v) {
                const int row = wm * 32 + mt * 16 + ((lane >> 4) << 3) + v;
                out[(size_t)(m0 + row) * 256 + col] = facc[mt][nt][v];
            }
        }
}

// ---------------------------------------------------------------------------
// Depthwise 3x3 + BN + residual ReLU (NHWC f32 -> NHWC f32)
// ---------------------------------------------------------------------------
__global__ __launch_bounds__(256)
void k_enhancer(const float* __restrict__ x0, const float* __restrict__ dww,
                const float* __restrict__ bn, float* __restrict__ x1,
                int H, int W, int total) {
    int idx = blockIdx.x * 256 + threadIdx.x;
    if (idx >= total) return;
    int c   = idx & 255;
    int pix = idx >> 8;
    int w   = pix % W;
    int tmp = pix / W;
    int h   = tmp % H;
    int b   = tmp / H;
    float acc = 0.0f;
#pragma unroll
    for (int ky = 0; ky < 3; ++ky)
#pragma unroll
        for (int kx = 0; kx < 3; ++kx) {
            int iy = h + ky - 1, ix = w + kx - 1;
            if (iy >= 0 && iy < H && ix >= 0 && ix < W)
                acc += x0[(size_t)((b * H + iy) * W + ix) * 256 + c] * dww[c * 9 + ky * 3 + kx];
        }
    float g = bn[c], be = bn[256 + c], mn = bn[512 + c], vr = bn[768 + c];
    float s = g * rsqrtf(vr + 1e-5f);
    float y = (acc - mn) * s + be + x0[idx];
    x1[idx] = fmaxf(y, 0.0f);
}

// 1x1 conv F=256 -> 32 (NHWC input), off stored [pix][32]
__global__ __launch_bounds__(256)
void k_offset_conv(const float* __restrict__ x, const float* __restrict__ w,
                   const float* __restrict__ bias, float* __restrict__ off, int total) {
    int idx = blockIdx.x * 256 + threadIdx.x;
    if (idx >= total) return;
    int oc  = idx & 31;
    int pix = idx >> 5;
    const float* xp = x + (size_t)pix * 256;
    const float* wp = w + oc * 256;
    float acc = bias[oc];
#pragma unroll 8
    for (int c = 0; c < 256; ++c) acc += xp[c] * wp[c];
    off[idx] = acc;
}

// ---------------------------------------------------------------------------
// DySample 2x (lp, groups=4): src = base_pixel + 0.25*off + (+-0.25),
// bilinear gather with border clamp. Input NHWC f32, output NHWC bf16.
// ---------------------------------------------------------------------------
__global__ __launch_bounds__(256)
void k_dysample(const float* __restrict__ x, const float* __restrict__ off,
                __bf16* __restrict__ out, int H, int W, int total) {
    int idx = blockIdx.x * 256 + threadIdx.x;
    if (idx >= total) return;
    int c   = idx & 255;
    int pix = idx >> 8;
    int Wo = W * 2, Ho = H * 2;
    int wo  = pix % Wo;
    int tmp = pix / Wo;
    int ho  = tmp % Ho;
    int b   = tmp / Ho;
    int h = ho >> 1, w = wo >> 1;
    int r1 = ho & 1, r2 = wo & 1;
    int g = c >> 6;
    int j = g * 4 + r1 * 2 + r2;
    const float* offp = off + (size_t)((b * H + h) * W + w) * 32;
    float sx = (float)w + 0.25f * offp[j]      + (r2 ? 0.25f : -0.25f);
    float sy = (float)h + 0.25f * offp[16 + j] + (r1 ? 0.25f : -0.25f);
    sx = fminf(fmaxf(sx, 0.0f), (float)(W - 1));
    sy = fminf(fmaxf(sy, 0.0f), (float)(H - 1));
    int x0i = (int)floorf(sx);
    int y0i = (int)floorf(sy);
    float wx = sx - (float)x0i;
    float wy = sy - (float)y0i;
    int x1i = (x0i + 1 < W) ? x0i + 1 : W - 1;
    int y1i = (y0i + 1 < H) ? y0i + 1 : H - 1;
    size_t r0 = (size_t)((b * H + y0i) * W) * 256 + c;
    size_t r1b = (size_t)((b * H + y1i) * W) * 256 + c;
    float v00 = x[r0  + (size_t)x0i * 256];
    float v01 = x[r0  + (size_t)x1i * 256];
    float v10 = x[r1b + (size_t)x0i * 256];
    float v11 = x[r1b + (size_t)x1i * 256];
    float v = v00 * (1.f - wy) * (1.f - wx) + v01 * (1.f - wy) * wx
            + v10 * wy * (1.f - wx)         + v11 * wy * wx;
    out[idx] = (__bf16)v;
}

// ---------------------------------------------------------------------------
// 3x3 conv, 256->256, implicit GEMM over K = 9*256 = 2304.
// Input NHWC bf16 (zero pad=1), weights [o][tap*256+i] bf16, BN+ReLU fused.
// Grid: (B*H*W/128, 2); 256 threads = 8 waves. Double-buffered async LDS.
// ---------------------------------------------------------------------------
__global__ __launch_bounds__(256)
void k_conv3x3_wmma(const __bf16* __restrict__ in, const __bf16* __restrict__ wt,
                    const float* __restrict__ bn, float* __restrict__ outF,
                    __bf16* __restrict__ outB, int H, int W) {
    __shared__ __bf16 Asl[2][128 * LDA];
    __shared__ __bf16 Bsl[2][128 * LDB];
    const int HW = H * W;
    const int m0 = blockIdx.x * 128;
    const int n0 = blockIdx.y * 128;
    const int t  = threadIdx.x;
    const int lane = t & 31;
    const int wv = t >> 5;
    const int wm = wv >> 1;
    const int wn = wv & 1;
    const int mm = t >> 1;
    const int kh = (t & 1) * 16;
    const int m  = m0 + mm;
    const int b  = m / HW;
    const int rem = m - b * HW;
    const int y  = rem / W;
    const int x  = rem - y * W;

    fx8 acc[2][4];
#pragma unroll
    for (int i = 0; i < 2; ++i)
#pragma unroll
        for (int j = 0; j < 4; ++j) acc[i][j] = zero_fx8();

    auto load_tile = [&](int kb, int buf) {
        const int tap = kb >> 3;
        const int c0  = (kb & 7) << 5;
        const int ky = tap / 3, kx = tap - ky * 3;
        const int iy = y + ky - 1, ix = x + kx - 1;
        __bf16* da = &Asl[buf][mm * LDA + kh];
        if (iy >= 0 && iy < H && ix >= 0 && ix < W) {
            const __bf16* src = in + ((size_t)(b * HW + iy * W + ix)) * 256 + c0 + kh;
            async_ld_b128(src,     da);
            async_ld_b128(src + 8, da + 8);
        } else {
            *(bfx8*)da       = zero_bfx8();
            *(bfx8*)(da + 8) = zero_bfx8();
        }
        const __bf16* sb = wt + (size_t)(n0 + mm) * 2304 + (kb << 5) + kh;
        __bf16* db = &Bsl[buf][mm * LDB + kh];
        async_ld_b128(sb,     db);
        async_ld_b128(sb + 8, db + 8);
    };

    load_tile(0, 0);
    wait_async0();
    __syncthreads();

    for (int kb = 0; kb < 72; ++kb) {
        const int cur = kb & 1;
        if (kb + 1 < 72) load_tile(kb + 1, 1 - cur);

        const int l16   = lane & 15;
        const int khalf = lane >> 4;
        bfx16 af[2], bfr[4];
#pragma unroll
        for (int mt = 0; mt < 2; ++mt) {
            const __bf16* p = &Asl[cur][(wm * 32 + mt * 16 + l16) * LDA];
            af[mt] = make_frag(p + khalf * 8, p + khalf * 8 + 16);
        }
#pragma unroll
        for (int nt = 0; nt < 4; ++nt) {
            const __bf16* p = &Bsl[cur][(wn * 64 + nt * 16 + l16) * LDB + khalf * 8];
            bfr[nt] = make_frag(p, p + 16);
        }
#pragma unroll
        for (int mt = 0; mt < 2; ++mt)
#pragma unroll
            for (int nt = 0; nt < 4; ++nt)
                acc[mt][nt] = __builtin_amdgcn_wmma_f32_16x16x32_bf16(
                    false, af[mt], false, bfr[nt], (short)0, acc[mt][nt],
                    false, false);
        wait_async0();
        __syncthreads();
    }

    // fused BN + ReLU epilogue
#pragma unroll
    for (int mt = 0; mt < 2; ++mt)
#pragma unroll
        for (int nt = 0; nt < 4; ++nt) {
            const int col = n0 + wn * 64 + nt * 16 + (lane & 15);
            float g = bn[col], be = bn[256 + col], mn = bn[512 + col], vr = bn[768 + col];
            float s = g * rsqrtf(vr + 1e-5f);
            float bb = be - mn * s;
#pragma unroll
            for (int v = 0; v < 8; ++v) {
                const int row = wm * 32 + mt * 16 + ((lane >> 4) << 3) + v;
                const size_t mg = (size_t)(m0 + row);
                float val = fmaxf(acc[mt][nt][v] * s + bb, 0.0f);
                if (outF) outF[mg * 256 + col] = val;
                if (outB) outB[mg * 256 + col] = (__bf16)val;
            }
        }
}

// ---------------------------------------------------------------------------
// Head: 3x3 conv 256 -> 1 (+bias). One wave per output pixel, wave32 reduce.
// ---------------------------------------------------------------------------
__global__ __launch_bounds__(256)
void k_head(const __bf16* __restrict__ y2, const float* __restrict__ w,
            const float* __restrict__ ob, float* __restrict__ out, int H, int W) {
    int gid  = blockIdx.x * 256 + threadIdx.x;
    int pix  = gid >> 5;
    int lane = gid & 31;
    int wo  = pix % W;
    int tmp = pix / W;
    int ho  = tmp % H;
    int b   = tmp / H;
    float acc = 0.0f;
#pragma unroll
    for (int ky = 0; ky < 3; ++ky)
#pragma unroll
        for (int kx = 0; kx < 3; ++kx) {
            int iy = ho + ky - 1, ix = wo + kx - 1;
            if (iy < 0 || iy >= H || ix < 0 || ix >= W) continue;
            const __bf16* row = y2 + (size_t)((b * H + iy) * W + ix) * 256;
            int tap = ky * 3 + kx;
#pragma unroll
            for (int cc = 0; cc < 256; cc += 32)
                acc += (float)row[cc + lane] * w[(cc + lane) * 9 + tap];
        }
#pragma unroll
    for (int o = 16; o > 0; o >>= 1) acc += __shfl_xor(acc, o, 32);
    if (lane == 0) out[pix] = acc + ob[0];
}

// ---------------------------------------------------------------------------
// Workspace layout (bytes, 256-aligned), total ~158 MB
// ---------------------------------------------------------------------------
static constexpr size_t OFF_FEATS = 0;          // [4][4][1024][768] bf16 : 50331648
static constexpr size_t OFF_PROJWT = 50331648;  // [4][256][768] bf16    : 1572864
static constexpr size_t OFF_X0   = 51904512;    // [4096][256] f32       : 4194304
static constexpr size_t OFF_X1   = 56098816;    // [4096][256] f32       : 4194304
static constexpr size_t OFF_OFF1 = 60293120;    // [4096][32] f32        : 524288
static constexpr size_t OFF_UP1  = 60817408;    // [16384][256] bf16     : 8388608
static constexpr size_t OFF_W1T  = 69206016;    // [256][2304] bf16      : 1179648
static constexpr size_t OFF_Y1   = 70385664;    // [16384][256] f32      : 16777216
static constexpr size_t OFF_OFF2 = 87162880;    // [16384][32] f32       : 2097152
static constexpr size_t OFF_UP2  = 89260032;    // [65536][256] bf16     : 33554432
static constexpr size_t OFF_W2T  = 122814464;   // [256][2304] bf16      : 1179648
static constexpr size_t OFF_Y2   = 123994112;   // [65536][256] bf16     : 33554432

extern "C" void kernel_launch(void* const* d_in, const int* in_sizes, int n_in,
                              void* d_out, int out_size, void* d_ws, size_t ws_size,
                              hipStream_t stream) {
    (void)in_sizes; (void)n_in; (void)out_size; (void)ws_size;
    const float* f[4] = {(const float*)d_in[0], (const float*)d_in[1],
                         (const float*)d_in[2], (const float*)d_in[3]};
    const float* proj_w  = (const float*)d_in[4];
    const float* layer_w = (const float*)d_in[5];
    const float* dw_w    = (const float*)d_in[6];
    const float* bn_enh  = (const float*)d_in[7];
    const float* ow1 = (const float*)d_in[8];
    const float* ob1 = (const float*)d_in[9];
    const float* cw1 = (const float*)d_in[10];
    const float* bn1 = (const float*)d_in[11];
    const float* ow2 = (const float*)d_in[12];
    const float* ob2 = (const float*)d_in[13];
    const float* cw2 = (const float*)d_in[14];
    const float* bn2 = (const float*)d_in[15];
    const float* out_w = (const float*)d_in[16];
    const float* out_b = (const float*)d_in[17];

    char* ws = (char*)d_ws;
    __bf16* FEATS = (__bf16*)(ws + OFF_FEATS);
    __bf16* PROJWT = (__bf16*)(ws + OFF_PROJWT);
    float*  X0   = (float*)(ws + OFF_X0);
    float*  X1   = (float*)(ws + OFF_X1);
    float*  OFF1 = (float*)(ws + OFF_OFF1);
    __bf16* UP1  = (__bf16*)(ws + OFF_UP1);
    __bf16* W1T  = (__bf16*)(ws + OFF_W1T);
    float*  Y1   = (float*)(ws + OFF_Y1);
    float*  OFF2 = (float*)(ws + OFF_OFF2);
    __bf16* UP2  = (__bf16*)(ws + OFF_UP2);
    __bf16* W2T  = (__bf16*)(ws + OFF_W2T);
    __bf16* Y2   = (__bf16*)(ws + OFF_Y2);

    // 1) feats NCHW f32 -> [l][b][n][c] bf16 ; proj_w -> [l][f][c] bf16
    for (int l = 0; l < 4; ++l)
        k_transpose_bf16<<<dim3(32, 24, 4), 256, 0, stream>>>(
            f[l], FEATS + (size_t)l * 4 * 1024 * 768, 768, 1024);
    for (int l = 0; l < 4; ++l)
        k_transpose_bf16<<<dim3(8, 24, 1), 256, 0, stream>>>(
            proj_w + (size_t)l * 768 * 256, PROJWT + (size_t)l * 256 * 768, 768, 256);
    k_reshape_convw<<<2304, 256, 0, stream>>>(cw1, W1T, 256 * 2304);
    k_reshape_convw<<<2304, 256, 0, stream>>>(cw2, W2T, 256 * 2304);

    // 2) weighted fusion (WMMA GEMM + GELU + softmax blend)
    k_fusion_wmma<<<dim3(32, 2), 256, 0, stream>>>(FEATS, PROJWT, layer_w, X0);

    // 3) spatial detail enhancer
    k_enhancer<<<4096, 256, 0, stream>>>(X0, dw_w, bn_enh, X1, 32, 32, 4 * 1024 * 256);

    // 4) dysample 1 + conv1 (WMMA) + BN + ReLU
    k_offset_conv<<<512, 256, 0, stream>>>(X1, ow1, ob1, OFF1, 4 * 1024 * 32);
    k_dysample<<<16384, 256, 0, stream>>>(X1, OFF1, UP1, 32, 32, 4 * 4096 * 256);
    k_conv3x3_wmma<<<dim3(128, 2), 256, 0, stream>>>(UP1, W1T, bn1, Y1, nullptr, 64, 64);

    // 5) dysample 2 + conv2 (WMMA) + BN + ReLU
    k_offset_conv<<<2048, 256, 0, stream>>>(Y1, ow2, ob2, OFF2, 4 * 4096 * 32);
    k_dysample<<<65536, 256, 0, stream>>>(Y1, OFF2, UP2, 64, 64, 4 * 16384 * 256);
    k_conv3x3_wmma<<<dim3(512, 2), 256, 0, stream>>>(UP2, W2T, bn2, nullptr, Y2, 128, 128);

    // 6) depth head
    k_head<<<8192, 256, 0, stream>>>(Y2, out_w, out_b, (float*)d_out, 128, 128);
}
